// Attention_2345052143845
// MI455X (gfx1250) — matrix operand order repaired and checked by reading, not compiled, not run on hardware
//
#include <hip/hip_runtime.h>

#define DEV_INLINE __device__ __forceinline__

typedef __attribute__((ext_vector_type(16))) _Float16 v16h;
typedef __attribute__((ext_vector_type(8)))  _Float16 v8h;
typedef __attribute__((ext_vector_type(8)))  float    v8f;

namespace {
constexpr int NB   = 32;    // batch
constexpr int C    = 384;   // channels (24 tiles = 6 strips of 4)
constexpr int TQ   = 785;   // q tokens (1 + 28*28)
constexpr int TQP  = 800;   // padded to /16
constexpr int TK   = 197;   // kv tokens (1 + 14*14)
constexpr int TKP  = 256;   // padded to /64 so every N dim is 4 tiles/wave exact
constexpr int NH   = 6;
constexpr int HD   = 64;
constexpr float SCALE_F = 0.05103103630798288f;  // 384^-0.5
}

union AFrag { v16h v; v8h h[2]; };

// A fragment (16x32 f16, row-major source): lane r=lane&15 holds row m0+r.
// Elements e -> K = k0 + (hi?16:0) + e (contiguous)
DEV_INLINE v16h load_a_frag(const _Float16* A, int lda, int row, int k0, int hi8) {
  const _Float16* p = A + (size_t)row * lda + k0 + hi8;
  AFrag f;
  f.h[0] = *(const v8h*)(p);
  f.h[1] = *(const v8h*)(p + 8);
  return f.v;
}

// B fragment (32x16 f16) from "B-transposed" storage [N,K] row-major:
// lane column n = lane&15, elements e -> K = k0 + (hi?16:0) + e (contiguous 32B)
DEV_INLINE v16h load_b_frag(const _Float16* Bm, int ldb, int col, int k0, int hi16) {
  const _Float16* p = Bm + (size_t)col * ldb + k0 + hi16;
  return *(const v16h*)p;
}

// Generic batched WMMA GEMM: D[M,N] = A[M,K] * B[N,K]^T, f32 accumulate.
// N MUST be a multiple of 64: each wave owns a 16x64 strip (4 N-tiles),
// unconditional inner loop -> no accumulator phi-copies.
template <bool OUTF32>
__global__ __launch_bounds__(128) void wmma_gemm(
    const _Float16* __restrict__ Abase, const _Float16* __restrict__ Bbase,
    void* __restrict__ Cbase,
    int M_tiles, int K,
    int lda, int ldb, int ldc,
    long long sAo, long long sAi, long long sBo, long long sBi,
    long long sCo, long long sCi, int inner_n,
    int Mreal, const float* __restrict__ bias)
{
  const int wave = threadIdx.x >> 5;
  const int lane = threadIdx.x & 31;
  const int mt = blockIdx.y * 4 + wave;
  if (mt >= M_tiles) return;            // wave-uniform

  const int z  = blockIdx.z;
  const int zo = z / inner_n;
  const int zi = z - zo * inner_n;

  const _Float16* A = Abase + zo * sAo + zi * sAi;
  const _Float16* B = Bbase + zo * sBo + zi * sBi;

  const int r    = lane & 15;
  const int hi8  = (lane >> 4) * 8;
  const int hi16 = (lane >> 4) * 16;
  const int m0   = mt * 16;
  const int n0   = blockIdx.x * 64;     // 4 consecutive N tiles

  v8f acc0 = v8f{0.f,0.f,0.f,0.f,0.f,0.f,0.f,0.f};
  v8f acc1 = acc0, acc2 = acc0, acc3 = acc0;

  const _Float16* b0 = B + (size_t)(n0 +  0 + r) * ldb + hi16;
  const _Float16* b1 = B + (size_t)(n0 + 16 + r) * ldb + hi16;
  const _Float16* b2 = B + (size_t)(n0 + 32 + r) * ldb + hi16;
  const _Float16* b3 = B + (size_t)(n0 + 48 + r) * ldb + hi16;

  for (int k0 = 0; k0 < K; k0 += 32) {
    v16h a = load_a_frag(A, lda, m0 + r, k0, hi16);
    v16h vb0 = *(const v16h*)(b0 + k0);
    v16h vb1 = *(const v16h*)(b1 + k0);
    v16h vb2 = *(const v16h*)(b2 + k0);
    v16h vb3 = *(const v16h*)(b3 + k0);
    acc0 = __builtin_amdgcn_wmma_f32_16x16x32_f16(false, a, false, vb0, (short)0, acc0, false, false);
    acc1 = __builtin_amdgcn_wmma_f32_16x16x32_f16(false, a, false, vb1, (short)0, acc1, false, false);
    acc2 = __builtin_amdgcn_wmma_f32_16x16x32_f16(false, a, false, vb2, (short)0, acc2, false, false);
    acc3 = __builtin_amdgcn_wmma_f32_16x16x32_f16(false, a, false, vb3, (short)0, acc3, false, false);
  }

  v8f accs[4] = {acc0, acc1, acc2, acc3};
#pragma unroll
  for (int nt = 0; nt < 4; ++nt) {
    const int col = n0 + nt * 16 + r;
    if (OUTF32) {
      float* Cf = (float*)Cbase + zo * sCo + zi * sCi;
      const float bv = bias ? bias[col] : 0.f;
#pragma unroll
      for (int j = 0; j < 8; ++j) {
        int row = m0 + j + hi8;
        if (row < Mreal) Cf[(size_t)row * ldc + col] = accs[nt][j] + bv;
      }
    } else {
      _Float16* Ch = (_Float16*)Cbase + zo * sCo + zi * sCi;
#pragma unroll
      for (int j = 0; j < 8; ++j) {
        int row = m0 + j + hi8;
        Ch[(size_t)row * ldc + col] = (_Float16)accs[nt][j];
      }
    }
  }
}

// Depthwise 3x3 conv + BN, cls-token passthrough at t==0, zero-fill pad rows.
__global__ void dwconv_bn_kernel(const float* __restrict__ x,
                                 const float* __restrict__ cw,
                                 const float* __restrict__ gamma,
                                 const float* __restrict__ beta,
                                 const float* __restrict__ mean,
                                 const float* __restrict__ var,
                                 _Float16* __restrict__ out,
                                 int which, int stride, int Wout, int Treal, int Tpad)
{
  int idx = blockIdx.x * blockDim.x + threadIdx.x;
  int total = NB * Tpad * C;
  if (idx >= total) return;
  int c = idx % C;
  int t = (idx / C) % Tpad;
  int b = idx / (C * Tpad);

  float r;
  if (t == 0) {
    r = x[(size_t)b * TQ * C + c];                     // cls token, no BN
  } else if (t < Treal) {
    int sp = t - 1;
    int oi = sp / Wout, oj = sp % Wout;
    const float* wc = cw + ((size_t)which * C + c) * 9;
    float acc = 0.f;
#pragma unroll
    for (int ky = 0; ky < 3; ++ky) {
      int ii = oi * stride + ky - 1;
      if (ii < 0 || ii >= 28) continue;
#pragma unroll
      for (int kx = 0; kx < 3; ++kx) {
        int jj = oj * stride + kx - 1;
        if (jj < 0 || jj >= 28) continue;
        acc += x[(size_t)b * TQ * C + (size_t)(1 + ii * 28 + jj) * C + c] * wc[ky * 3 + kx];
      }
    }
    float inv = gamma[which * C + c] * rsqrtf(var[which * C + c] + 1e-5f);
    r = acc * inv + (beta[which * C + c] - mean[which * C + c] * inv);
  } else {
    r = 0.f;                                           // pad rows -> zero
  }
  out[(size_t)b * Tpad * C + (size_t)t * C + c] = (_Float16)r;
}

__global__ void convert_weights(const float* __restrict__ wq, const float* __restrict__ wk,
                                const float* __restrict__ wv, const float* __restrict__ wp,
                                _Float16* __restrict__ out, int n)
{
  int i = blockIdx.x * blockDim.x + threadIdx.x;
  if (i >= n) return;
  int which = i / (C * C), rr = i % (C * C);
  const float* src = which == 0 ? wq : which == 1 ? wk : which == 2 ? wv : wp;
  out[i] = (_Float16)src[rr];
}

// V [B, TKP, C] -> Vt [B, NH, HD, TKP] so P@V B-fragments load contiguously.
__global__ void transpose_v(const _Float16* __restrict__ Vp, _Float16* __restrict__ Vt, int n)
{
  int idx = blockIdx.x * blockDim.x + threadIdx.x;
  if (idx >= n) return;
  int t  = idx % TKP;
  int d  = (idx / TKP) % HD;
  int hh = (idx / (TKP * HD)) % NH;
  int b  = idx / (TKP * HD * NH);
  Vt[idx] = Vp[(size_t)b * TKP * C + (size_t)t * C + hh * HD + d];
}

// One wave per score row: scale, masked softmax over 197 real cols, zero pads.
__global__ void softmax_rows(_Float16* __restrict__ S, int nrows)
{
  int wid  = (blockIdx.x * blockDim.x + threadIdx.x) >> 5;
  int lane = threadIdx.x & 31;
  if (wid >= nrows) return;
  _Float16* row = S + (size_t)wid * TKP;

  float v[TKP / 32];
  float mx = -3.0e30f;
#pragma unroll
  for (int i = 0; i < TKP / 32; ++i) {
    int t = lane + 32 * i;
    float s = (t < TK) ? (float)row[t] * SCALE_F : -3.0e30f;
    v[i] = s;
    mx = fmaxf(mx, s);
  }
#pragma unroll
  for (int off = 16; off > 0; off >>= 1) mx = fmaxf(mx, __shfl_xor(mx, off, 32));

  float sum = 0.f;
#pragma unroll
  for (int i = 0; i < TKP / 32; ++i) {
    float e = __expf(v[i] - mx);   // pads underflow to 0
    v[i] = e;
    sum += e;
  }
#pragma unroll
  for (int off = 16; off > 0; off >>= 1) sum += __shfl_xor(sum, off, 32);

  float rinv = 1.f / sum;
#pragma unroll
  for (int i = 0; i < TKP / 32; ++i) {
    int t = lane + 32 * i;
    row[t] = (_Float16)(v[i] * rinv);
  }
}

extern "C" void kernel_launch(void* const* d_in, const int* in_sizes, int n_in,
                              void* d_out, int out_size, void* d_ws, size_t ws_size,
                              hipStream_t stream)
{
  const float* x      = (const float*)d_in[0];
  const float* conv_w = (const float*)d_in[1];
  const float* gm     = (const float*)d_in[2];
  const float* bt     = (const float*)d_in[3];
  const float* mn     = (const float*)d_in[4];
  const float* vr     = (const float*)d_in[5];
  const float* wq     = (const float*)d_in[6];
  const float* wk     = (const float*)d_in[7];
  const float* wv     = (const float*)d_in[8];
  const float* wp     = (const float*)d_in[9];
  const float* bp     = (const float*)d_in[10];
  float* out = (float*)d_out;
  (void)in_sizes; (void)n_in; (void)out_size; (void)ws_size;

  char* ws = (char*)d_ws;
  size_t off = 0;
  auto alloc = [&](size_t bytes) -> char* {
    char* p = ws + off;
    off = (off + bytes + 255) & ~(size_t)255;
    return p;
  };
  _Float16* wh = (_Float16*)alloc((size_t)4 * C * C * 2);          // f16 weights
  _Float16* qh = (_Float16*)alloc((size_t)NB * TQP * C * 2);       // conv outputs
  _Float16* kh = (_Float16*)alloc((size_t)NB * TKP * C * 2);
  _Float16* vh = (_Float16*)alloc((size_t)NB * TKP * C * 2);
  _Float16* Qp = (_Float16*)alloc((size_t)NB * TQP * C * 2);       // projections
  _Float16* Kp = (_Float16*)alloc((size_t)NB * TKP * C * 2);
  _Float16* Vp = (_Float16*)alloc((size_t)NB * TKP * C * 2);
  _Float16* Vt = (_Float16*)alloc((size_t)NB * NH * HD * TKP * 2); // V transposed per head
  _Float16* S  = (_Float16*)alloc((size_t)NB * NH * TQP * TKP * 2);// scores / probs
  _Float16* AO = (_Float16*)alloc((size_t)NB * TQP * C * 2);       // attn output (heads concat)

  dim3 blk(128, 1, 1);  // 4 waves/block, one 16x64 strip per wave

  { int n = 4 * C * C;
    convert_weights<<<(n + 255) / 256, 256, 0, stream>>>(wq, wk, wv, wp, wh, n); }

  { int n = NB * TQP * C;
    dwconv_bn_kernel<<<(n + 255) / 256, 256, 0, stream>>>(x, conv_w, gm, bt, mn, vr, qh, 0, 1, 28, TQ, TQP); }
  { int n = NB * TKP * C;
    dwconv_bn_kernel<<<(n + 255) / 256, 256, 0, stream>>>(x, conv_w, gm, bt, mn, vr, kh, 1, 2, 14, TK, TKP); }
  { int n = NB * TKP * C;
    dwconv_bn_kernel<<<(n + 255) / 256, 256, 0, stream>>>(x, conv_w, gm, bt, mn, vr, vh, 2, 2, 14, TK, TKP); }

  // Q/K/V projections: out = act @ W^T   (W stored [N=C, K=C]); N = 384 = 6 strips
  wmma_gemm<false><<<dim3(6, 13, NB), blk, 0, stream>>>(
      qh, wh + 0 * C * C, Qp, TQP / 16, C, C, C, C,
      (long long)TQP * C, 0, 0, 0, (long long)TQP * C, 0, 1, 0, nullptr);
  wmma_gemm<false><<<dim3(6, 4, NB), blk, 0, stream>>>(
      kh, wh + 1 * C * C, Kp, TKP / 16, C, C, C, C,
      (long long)TKP * C, 0, 0, 0, (long long)TKP * C, 0, 1, 0, nullptr);
  wmma_gemm<false><<<dim3(6, 4, NB), blk, 0, stream>>>(
      vh, wh + 2 * C * C, Vp, TKP / 16, C, C, C, C,
      (long long)TKP * C, 0, 0, 0, (long long)TKP * C, 0, 1, 0, nullptr);

  // S = Q @ K^T per (b, h): N = TKP = 256 = 4 strips, K = HD
  wmma_gemm<false><<<dim3(TKP / 64, 13, NB * NH), blk, 0, stream>>>(
      Qp, Kp, S, TQP / 16, HD, C, C, TKP,
      (long long)TQP * C, HD, (long long)TKP * C, HD,
      (long long)NH * TQP * TKP, (long long)TQP * TKP, NH, 0, nullptr);

  { int rows = NB * NH * TQP;
    softmax_rows<<<(rows * 32 + 255) / 256, 256, 0, stream>>>(S, rows); }

  { int n = NB * NH * HD * TKP;
    transpose_v<<<(n + 255) / 256, 256, 0, stream>>>(Vp, Vt, n); }

  // O = P @ V per (b, h): N = HD = 64 = 1 strip, K = TKP
  wmma_gemm<false><<<dim3(1, 13, NB * NH), blk, 0, stream>>>(
      S, Vt, AO, TQP / 16, TKP, TKP, TKP, C,
      (long long)NH * TQP * TKP, (long long)TQP * TKP,
      (long long)NH * HD * TKP, (long long)HD * TKP,
      (long long)TQP * C, HD, NH, 0, nullptr);

  // Final projection + bias, f32 output, guard real rows (785)
  wmma_gemm<true><<<dim3(6, 13, NB), blk, 0, stream>>>(
      AO, wh + 3 * C * C, out, TQP / 16, C, C, C, C,
      (long long)TQP * C, 0, 0, 0, (long long)TQ * C, 0, 1, TQ, bp);
}
